// ConflictModel_26250840113730
// MI455X (gfx1250) — matrix operand order repaired and checked by reading, not compiled, not run on hardware
//
#include <hip/hip_runtime.h>

typedef _Float16 f16;
typedef __attribute__((ext_vector_type(16))) _Float16 v16h;
typedef __attribute__((ext_vector_type(8)))  _Float16 v8h;
typedef __attribute__((ext_vector_type(8)))  float    v8f;

#define A_DIM   128
#define E_DIM   256
#define N_CITY  512
#define H_DIM   32
#define NHEAD   8
#define HID_DIM 1024
#define NLAYER  3
#define NEGV    (-1e9f)

// ---------------- workspace layout (units: halves) ----------------
#define WS_WQ 0
#define WS_WK (WS_WQ + NLAYER*E_DIM*E_DIM)
#define WS_WV (WS_WK + NLAYER*E_DIM*E_DIM)
#define WS_WO (WS_WV + NLAYER*E_DIM*E_DIM)
#define WS_W1 (WS_WO + NLAYER*E_DIM*E_DIM)      // [l][HID][E]   (transposed)
#define WS_W2 (WS_W1 + NLAYER*E_DIM*HID_DIM)    // [l][E][HID]   (transposed)
#define WS_SQ (WS_W2 + NLAYER*HID_DIM*E_DIM)    // [E][E]        (transposed)
#define WS_SK (WS_SQ + E_DIM*E_DIM)

// ---------------- LDS layout (bytes) ----------------
#define OFF_CAC   0        // f16 [128][256]   residual stream (queries)
#define OFF_KV    65536    // f16 [128][256]   agent_embed (keys/values src)
#define OFF_AOUT  131072   // f16 [128][256]   attn out | preLN2 | final q
#define OFF_Q     196608   // f16 [128][32]
#define OFF_K     204800   // f16 [128][32]
#define OFF_VT    212992   // f16 [32][128]    v transposed
#define OFF_SC    221184   // f32 [128][128] scores | f16 preLN1 | f16 h1-chunk | f16 final k
#define OFF_ATT   286720   // f16 [128][128]   unnormalized softmax
#define OFF_ACTS  319488   // i32 [128]
#define OFF_INV   320000   // f32 [128]        1/rowsum
#define SMEM_SZ   320512

// ------------- WMMA fragment loaders (16-bit, 16x16x32) -------------
// A-matrix 16x32: lane m=lane&15; lanes<16 -> K {0..7,16..23}; lanes>=16 -> K {8..15,24..31}
__device__ inline v16h ld_a(const f16* p, int ld, int lane) {
  const f16* r = p + (lane & 15) * ld + ((lane >> 4) << 3);
  v8h lo = *(const v8h*)(r);
  v8h hi = *(const v8h*)(r + 16);
  v16h f;
#pragma unroll
  for (int i = 0; i < 8; ++i) { f[i] = lo[i]; f[i + 8] = hi[i]; }
  return f;
}
// B-matrix 32x16 supplied as Bt[n][k] (row-major, n=lane&15):
// lanes<16 -> K {0..7,16..23}; lanes>=16 -> K {8..15,24..31}.
__device__ inline v16h ld_b(const f16* p, int ld, int lane) {
  const f16* r = p + (lane & 15) * ld + ((lane >> 4) << 3);
  v8h lo = *(const v8h*)(r);
  v8h hi = *(const v8h*)(r + 16);
  v16h f;
#pragma unroll
  for (int i = 0; i < 8; ++i) { f[i] = lo[i]; f[i + 8] = hi[i]; }
  return f;
}

// Generic M=128 GEMM: C[128][N] = A[128][K] * Bt[N][K]^T, tiles cycled over 8 waves.
template <typename EmitF>
__device__ inline void gemm128(const f16* As, int lda, const f16* Bt, int ldb,
                               int N, int K, int wave, int lane, EmitF emit) {
  const int tn_cnt = N >> 4;
  const int ntiles = (A_DIM >> 4) * tn_cnt;
  for (int t = wave; t < ntiles; t += 8) {
    const int tm = t / tn_cnt, tn = t % tn_cnt;
    v8f acc = {};
    for (int kk = 0; kk < K; kk += 32) {
      v16h a = ld_a(As + tm * 16 * lda + kk, lda, lane);
      v16h b = ld_b(Bt + tn * 16 * ldb + kk, ldb, lane);
      acc = __builtin_amdgcn_wmma_f32_16x16x32_f16(false, a, false, b,
                                                   (short)0, acc, false, false);
    }
    emit(tm, tn, acc);
  }
}

// ------------- weight convert: f32 [l][R][C] -> f16 [l][C][R] -------------
__global__ void cvt_transpose_f16(const float* __restrict__ src, f16* __restrict__ dst,
                                  int L, int R, int C) {
  const long total = (long)L * R * C;
  for (long i = (long)blockIdx.x * blockDim.x + threadIdx.x; i < total;
       i += (long)gridDim.x * blockDim.x) {
    int l = (int)(i / ((long)R * C));
    int rem = (int)(i - (long)l * R * C);
    int r = rem / C, c = rem % C;
    dst[((long)l * C + c) * R + r] = (f16)src[i];
  }
}

// ------------- main fused transformer kernel: 1 block per batch element -------------
__global__ __launch_bounds__(256)
void conflict_model_kernel(const float* __restrict__ agent_embed,
                           const float* __restrict__ city_embed,
                           const int*   __restrict__ acts,
                           const float* __restrict__ ffn_b1,
                           const float* __restrict__ ffn_b2,
                           const float* __restrict__ ln1_g, const float* __restrict__ ln1_b,
                           const float* __restrict__ ln2_g, const float* __restrict__ ln2_b,
                           const f16*   __restrict__ ws,
                           float*       __restrict__ out) {
  extern __shared__ char smem[];
  f16*   s_cac  = (f16*)(smem + OFF_CAC);
  f16*   s_kv   = (f16*)(smem + OFF_KV);
  f16*   s_aout = (f16*)(smem + OFF_AOUT);
  f16*   s_q    = (f16*)(smem + OFF_Q);
  f16*   s_k    = (f16*)(smem + OFF_K);
  f16*   s_vt   = (f16*)(smem + OFF_VT);
  float* s_sc   = (float*)(smem + OFF_SC);
  f16*   s_att  = (f16*)(smem + OFF_ATT);
  int*   s_acts = (int*)(smem + OFF_ACTS);
  float* s_inv  = (float*)(smem + OFF_INV);

  const int b = blockIdx.x;
  const int tid = threadIdx.x;
  const int wave = tid >> 5, lane = tid & 31;

  if (tid < A_DIM) s_acts[tid] = acts[(size_t)b * A_DIM + tid];
  __syncthreads();

  // gather chosen-city embeddings (queries) and agent embeddings (keys/values), cvt to f16
  {
    const int total4 = A_DIM * E_DIM / 4;
    for (int i = tid; i < total4; i += 256) {
      const int m = i >> 6;
      const int c = (i & 63) << 2;
      float4 cv = *(const float4*)(city_embed + ((size_t)b * N_CITY + s_acts[m]) * E_DIM + c);
      f16* d = s_cac + m * E_DIM + c;
      d[0] = (f16)cv.x; d[1] = (f16)cv.y; d[2] = (f16)cv.z; d[3] = (f16)cv.w;
      float4 av = *(const float4*)(agent_embed + ((size_t)b * A_DIM + m) * E_DIM + c);
      f16* d2 = s_kv + m * E_DIM + c;
      d2[0] = (f16)av.x; d2[1] = (f16)av.y; d2[2] = (f16)av.z; d2[3] = (f16)av.w;
    }
  }
  __syncthreads();

  const float att_scale = 0.17677669529663687f; // 1/sqrt(32)

  for (int l = 0; l < NLAYER; ++l) {
    const f16* wq = ws + WS_WQ + l * E_DIM * E_DIM;
    const f16* wk = ws + WS_WK + l * E_DIM * E_DIM;
    const f16* wv = ws + WS_WV + l * E_DIM * E_DIM;
    const f16* wo = ws + WS_WO + l * E_DIM * E_DIM;
    const f16* w1 = ws + WS_W1 + l * E_DIM * HID_DIM; // [HID][E]
    const f16* w2 = ws + WS_W2 + l * HID_DIM * E_DIM; // [E][HID]

    // ---- multi-head cross attention (q from cac, k/v from agent_embed) ----
    for (int h = 0; h < NHEAD; ++h) {
      gemm128(s_cac, E_DIM, wq + (h * H_DIM) * E_DIM, E_DIM, H_DIM, E_DIM, wave, lane,
        [&](int tm, int tn, v8f acc) {
          const int n = lane & 15, mb = (lane >> 4) * 8;
#pragma unroll
          for (int r = 0; r < 8; ++r)
            s_q[(tm * 16 + mb + r) * H_DIM + tn * 16 + n] = (f16)acc[r];
        });
      gemm128(s_kv, E_DIM, wk + (h * H_DIM) * E_DIM, E_DIM, H_DIM, E_DIM, wave, lane,
        [&](int tm, int tn, v8f acc) {
          const int n = lane & 15, mb = (lane >> 4) * 8;
#pragma unroll
          for (int r = 0; r < 8; ++r)
            s_k[(tm * 16 + mb + r) * H_DIM + tn * 16 + n] = (f16)acc[r];
        });
      gemm128(s_kv, E_DIM, wv + (h * H_DIM) * E_DIM, E_DIM, H_DIM, E_DIM, wave, lane,
        [&](int tm, int tn, v8f acc) { // store transposed -> direct Bt operand
          const int n = lane & 15, mb = (lane >> 4) * 8;
#pragma unroll
          for (int r = 0; r < 8; ++r)
            s_vt[(tn * 16 + n) * A_DIM + tm * 16 + mb + r] = (f16)acc[r];
        });
      __syncthreads();

      // scores = q k^T * scale, conflict-masked
      gemm128(s_q, H_DIM, s_k, H_DIM, A_DIM, H_DIM, wave, lane,
        [&](int tm, int tn, v8f acc) {
          const int n = lane & 15, mb = (lane >> 4) * 8;
          const int j = tn * 16 + n;
          const int aj = s_acts[j];
#pragma unroll
          for (int r = 0; r < 8; ++r) {
            const int i = tm * 16 + mb + r;
            s_sc[i * A_DIM + j] = (s_acts[i] == aj) ? acc[r] * att_scale : NEGV;
          }
        });
      __syncthreads();

      // row softmax: store unnormalized exp (f16); fold 1/sum into attn@v emit
      if (tid < A_DIM) {
        const float* row = s_sc + tid * A_DIM;
        float mx = row[0];
        for (int j = 1; j < A_DIM; ++j) mx = fmaxf(mx, row[j]);
        float sum = 0.f;
        for (int j = 0; j < A_DIM; ++j) {
          float e = __expf(row[j] - mx);
          sum += e;
          s_att[tid * A_DIM + j] = (f16)e;
        }
        s_inv[tid] = 1.f / sum;
      }
      __syncthreads();

      // out_h = softmax(scores) @ v  -> columns [h*32, h*32+32) of attn_out
      gemm128(s_att, A_DIM, s_vt, A_DIM, H_DIM, A_DIM, wave, lane,
        [&](int tm, int tn, v8f acc) {
          const int n = lane & 15, mb = (lane >> 4) * 8;
#pragma unroll
          for (int r = 0; r < 8; ++r) {
            const int m = tm * 16 + mb + r;
            s_aout[m * E_DIM + h * H_DIM + tn * 16 + n] = (f16)(acc[r] * s_inv[m]);
          }
        });
      __syncthreads();
    }

    // ---- output projection + residual -> preLN1 (reuse scores region) ----
    f16* s_pre = (f16*)(smem + OFF_SC);
    gemm128(s_aout, E_DIM, wo, E_DIM, E_DIM, E_DIM, wave, lane,
      [&](int tm, int tn, v8f acc) {
        const int n = lane & 15, mb = (lane >> 4) * 8;
        const int col = tn * 16 + n;
#pragma unroll
        for (int r = 0; r < 8; ++r) {
          const int m = tm * 16 + mb + r;
          s_pre[m * E_DIM + col] = (f16)(acc[r] + (float)s_cac[m * E_DIM + col]);
        }
      });
    __syncthreads();

    if (tid < A_DIM) { // LayerNorm 1 -> cac
      const f16* row = s_pre + tid * E_DIM;
      float mean = 0.f;
      for (int c = 0; c < E_DIM; ++c) mean += (float)row[c];
      mean *= (1.f / E_DIM);
      float var = 0.f;
      for (int c = 0; c < E_DIM; ++c) { float d = (float)row[c] - mean; var += d * d; }
      var *= (1.f / E_DIM);
      const float rs = rsqrtf(var + 1e-5f);
      for (int c = 0; c < E_DIM; ++c)
        s_cac[tid * E_DIM + c] =
            (f16)(((float)row[c] - mean) * rs * ln1_g[l * E_DIM + c] + ln1_b[l * E_DIM + c]);
    }
    __syncthreads();

    // ---- FFN: 256 -> 1024 (relu) -> 256, hidden streamed in 4 chunks ----
    f16* s_h1 = (f16*)(smem + OFF_SC);
    v8f acc2[16];
#pragma unroll
    for (int i = 0; i < 16; ++i) acc2[i] = (v8f){};
    for (int c = 0; c < 4; ++c) {
      gemm128(s_cac, E_DIM, w1 + (c * E_DIM) * E_DIM, E_DIM, E_DIM, E_DIM, wave, lane,
        [&](int tm, int tn, v8f acc) {
          const int n = lane & 15, mb = (lane >> 4) * 8;
          const int col = tn * 16 + n;
          const float bias = ffn_b1[l * HID_DIM + c * E_DIM + col];
#pragma unroll
          for (int r = 0; r < 8; ++r) {
            float v = fmaxf(acc[r] + bias, 0.f);
            s_h1[(tm * 16 + mb + r) * E_DIM + col] = (f16)v;
          }
        });
      __syncthreads();
#pragma unroll
      for (int i = 0; i < 16; ++i) { // accumulate h1_chunk @ W2 across chunks
        const int t = wave + 8 * i;
        const int tm = t >> 4, tn = t & 15;
        for (int kk = 0; kk < E_DIM; kk += 32) {
          v16h a = ld_a(s_h1 + tm * 16 * E_DIM + kk, E_DIM, lane);
          v16h bb = ld_b(w2 + tn * 16 * HID_DIM + c * E_DIM + kk, HID_DIM, lane);
          acc2[i] = __builtin_amdgcn_wmma_f32_16x16x32_f16(false, a, false, bb,
                                                           (short)0, acc2[i], false, false);
        }
      }
      __syncthreads();
    }
    f16* s_pre2 = (f16*)(smem + OFF_AOUT);
#pragma unroll
    for (int i = 0; i < 16; ++i) { // + bias + residual -> preLN2
      const int t = wave + 8 * i;
      const int tm = t >> 4, tn = t & 15;
      const int n = lane & 15, mb = (lane >> 4) * 8;
      const int col = tn * 16 + n;
      const float bias = ffn_b2[l * E_DIM + col];
#pragma unroll
      for (int r = 0; r < 8; ++r) {
        const int m = tm * 16 + mb + r;
        s_pre2[m * E_DIM + col] = (f16)(acc2[i][r] + bias + (float)s_cac[m * E_DIM + col]);
      }
    }
    __syncthreads();

    if (tid < A_DIM) { // LayerNorm 2 -> cac
      const f16* row = s_pre2 + tid * E_DIM;
      float mean = 0.f;
      for (int c = 0; c < E_DIM; ++c) mean += (float)row[c];
      mean *= (1.f / E_DIM);
      float var = 0.f;
      for (int c = 0; c < E_DIM; ++c) { float d = (float)row[c] - mean; var += d * d; }
      var *= (1.f / E_DIM);
      const float rs = rsqrtf(var + 1e-5f);
      for (int c = 0; c < E_DIM; ++c)
        s_cac[tid * E_DIM + c] =
            (f16)(((float)row[c] - mean) * rs * ln2_g[l * E_DIM + c] + ln2_b[l * E_DIM + c]);
    }
    __syncthreads();
  }

  // ---- final single-head pointer logits ----
  f16* s_qf = (f16*)(smem + OFF_AOUT);
  f16* s_kf = (f16*)(smem + OFF_SC);
  gemm128(s_cac, E_DIM, ws + WS_SQ, E_DIM, E_DIM, E_DIM, wave, lane,
    [&](int tm, int tn, v8f acc) {
      const int n = lane & 15, mb = (lane >> 4) * 8;
#pragma unroll
      for (int r = 0; r < 8; ++r)
        s_qf[(tm * 16 + mb + r) * E_DIM + tn * 16 + n] = (f16)acc[r];
    });
  gemm128(s_kv, E_DIM, ws + WS_SK, E_DIM, E_DIM, E_DIM, wave, lane,
    [&](int tm, int tn, v8f acc) {
      const int n = lane & 15, mb = (lane >> 4) * 8;
#pragma unroll
      for (int r = 0; r < 8; ++r)
        s_kf[(tm * 16 + mb + r) * E_DIM + tn * 16 + n] = (f16)acc[r];
    });
  __syncthreads();

  const float inv_sqrt_e = 1.f / 16.f;
  gemm128(s_qf, E_DIM, s_kf, E_DIM, A_DIM, E_DIM, wave, lane,
    [&](int tm, int tn, v8f acc) {
      const int n = lane & 15, mb = (lane >> 4) * 8;
      const int j = tn * 16 + n;
      const int aj = s_acts[j];
#pragma unroll
      for (int r = 0; r < 8; ++r) {
        const int i = tm * 16 + mb + r;
        float v = 10.f * tanhf(acc[r] * inv_sqrt_e);
        out[((size_t)b * A_DIM + i) * A_DIM + j] = (s_acts[i] == aj) ? v : NEGV;
      }
    });
}

// ---------------- host side ----------------
static void launch_cvt(const float* src, f16* dst, int L, int R, int C, hipStream_t s) {
  long total = (long)L * R * C;
  int blocks = (int)((total + 255) / 256);
  if (blocks > 4096) blocks = 4096;
  cvt_transpose_f16<<<blocks, 256, 0, s>>>(src, dst, L, R, C);
}

extern "C" void kernel_launch(void* const* d_in, const int* in_sizes, int n_in,
                              void* d_out, int out_size, void* d_ws, size_t ws_size,
                              hipStream_t stream) {
  const float* agent = (const float*)d_in[0];
  const float* city  = (const float*)d_in[1];
  const int*   acts  = (const int*)d_in[2];
  const float* Wq = (const float*)d_in[3];
  const float* Wk = (const float*)d_in[4];
  const float* Wv = (const float*)d_in[5];
  const float* Wo = (const float*)d_in[6];
  const float* W1 = (const float*)d_in[7];
  const float* b1 = (const float*)d_in[8];
  const float* W2 = (const float*)d_in[9];
  const float* b2 = (const float*)d_in[10];
  const float* g1 = (const float*)d_in[11];
  const float* bb1 = (const float*)d_in[12];
  const float* g2 = (const float*)d_in[13];
  const float* bb2 = (const float*)d_in[14];
  const float* shaq = (const float*)d_in[15];
  const float* shak = (const float*)d_in[16];
  f16* ws = (f16*)d_ws;
  const int B = in_sizes[0] / (A_DIM * E_DIM);

  // one-shot weight convert+transpose to f16 (stays hot in 192MB L2)
  launch_cvt(Wq, ws + WS_WQ, NLAYER, E_DIM, E_DIM, stream);
  launch_cvt(Wk, ws + WS_WK, NLAYER, E_DIM, E_DIM, stream);
  launch_cvt(Wv, ws + WS_WV, NLAYER, E_DIM, E_DIM, stream);
  launch_cvt(Wo, ws + WS_WO, NLAYER, E_DIM, E_DIM, stream);
  launch_cvt(W1, ws + WS_W1, NLAYER, E_DIM, HID_DIM, stream);
  launch_cvt(W2, ws + WS_W2, NLAYER, HID_DIM, E_DIM, stream);
  launch_cvt(shaq, ws + WS_SQ, 1, E_DIM, E_DIM, stream);
  launch_cvt(shak, ws + WS_SK, 1, E_DIM, E_DIM, stream);

  (void)hipFuncSetAttribute((const void*)conflict_model_kernel,
                            hipFuncAttributeMaxDynamicSharedMemorySize, SMEM_SZ);
  conflict_model_kernel<<<dim3(B), dim3(256), SMEM_SZ, stream>>>(
      agent, city, acts, b1, b2, g1, bb1, g2, bb2, (const f16*)ws, (float*)d_out);
  (void)n_in; (void)out_size; (void)ws_size;
}
